// MlpwithSOMModule_34720515621527
// MI455X (gfx1250) — matrix operand order repaired and checked by reading, not compiled, not run on hardware
//
#include <hip/hip_runtime.h>
#include <hip/hip_bf16.h>

// MI455X / gfx1250, wave32. Matmuls on v_wmma_f32_16x16x32_bf16.
// Workspace: X   = bf16[65536][768]  gathered pair matrix (96 MB, L2-resident)
//            H1  = bf16[65536][768]  layer-1 activations  (96 MB, L2-resident)
//            Wt1 = bf16[768][768]    W1 transposed [n][k] (1.125 MB)
//            Wt2 = bf16[768][768]    W2 transposed [n][k] (1.125 MB)

typedef __attribute__((ext_vector_type(16))) __bf16 v16bf;
typedef __attribute__((ext_vector_type(8)))  float  v8f;
typedef int i4v __attribute__((vector_size(16)));

#define DDIM 768
#define LDIM 128
#define NBK  256          // B*K
#define CH   64           // K-chunk staged in LDS
#define NCH  (DDIM / CH)  // 12
#define LSTR 72           // LDS row stride (shorts): 144B, 16B-aligned, conflict-free

#define GAS __attribute__((address_space(1)))
#define LAS __attribute__((address_space(3)))

#if __has_builtin(__builtin_amdgcn_global_load_async_to_lds_b128)
#define HAVE_ASYNC_LDS 1
#endif

// 16-byte global -> LDS copy; async (ASYNCcnt) when the builtin exists.
__device__ __forceinline__ void cp16(void* lds, const void* g) {
#ifdef HAVE_ASYNC_LDS
  __builtin_amdgcn_global_load_async_to_lds_b128(
      (GAS i4v*)(unsigned long long)(g),
      (LAS i4v*)(unsigned int)(unsigned long long)(lds), 0, 0);
#else
  *(uint4*)lds = *(const uint4*)g;
#endif
}

__device__ __forceinline__ void wait_async0() {
#ifdef HAVE_ASYNC_LDS
#if __has_builtin(__builtin_amdgcn_s_wait_asynccnt)
  __builtin_amdgcn_s_wait_asynccnt(0);
#else
  asm volatile("s_wait_asynccnt 0x0" ::: "memory");
#endif
#endif
}

__device__ __forceinline__ unsigned short f2bf(float f) {
  union { float f; unsigned u; } x; x.f = f;
  unsigned r = x.u + 0x7FFFu + ((x.u >> 16) & 1u);   // round-to-nearest-even
  return (unsigned short)(r >> 16);
}

__device__ __forceinline__ v8f v8f_zero() {
  v8f z;
#pragma unroll
  for (int i = 0; i < 8; ++i) z[i] = 0.0f;
  return z;
}

// A fragment (16x32 bf16, MxK): lanes 0-15 hold K=0..7 & 16..23, lanes 16-31 hold K=8..15 & 24..31
__device__ __forceinline__ v16bf frag_a(const unsigned short* s, int row, int kk, int lane) {
  const int kb = (lane >> 4) << 3;
  v16bf a;
  ((uint4*)&a)[0] = *(const uint4*)(s + row * LSTR + kk + kb);
  ((uint4*)&a)[1] = *(const uint4*)(s + row * LSTR + kk + kb + 16);
  return a;
}
// B fragment (32x16 bf16, KxN) from an [N][K]-major LDS tile: lane=N, lanes 0-15 hold K=0..7 & 16..23, lanes 16-31 hold K=8..15 & 24..31
__device__ __forceinline__ v16bf frag_b(const unsigned short* s, int nrow, int kk, int lane) {
  const int kb = (lane >> 4) << 3;
  v16bf b;
  ((uint4*)&b)[0] = *(const uint4*)(s + nrow * LSTR + kk + kb);
  ((uint4*)&b)[1] = *(const uint4*)(s + nrow * LSTR + kk + kb + 16);
  return b;
}

// ---------------- Kernel 0: Wt[n][k] = bf16(W[k][n]) (one-time convert+transpose) --------
__global__ void __launch_bounds__(256)
k_wtrans(const float* __restrict__ W, unsigned short* __restrict__ Wt) {
  __shared__ unsigned short tile[64][72];
  const int bx = blockIdx.x % 12, by = blockIdx.x / 12;
  const int k0 = bx * 64, n0 = by * 64;
  const int t = threadIdx.x;
#pragma unroll
  for (int q = 0; q < 4; ++q) {
    int i = q * 256 + t;                 // 1024 float4 over 64x64
    int r = i >> 4, c4 = i & 15;         // r = k-row, c4 = n-group
    float4 v = *(const float4*)(W + (size_t)(k0 + r) * DDIM + n0 + c4 * 4);
    tile[c4 * 4 + 0][r] = f2bf(v.x);
    tile[c4 * 4 + 1][r] = f2bf(v.y);
    tile[c4 * 4 + 2][r] = f2bf(v.z);
    tile[c4 * 4 + 3][r] = f2bf(v.w);
  }
  __syncthreads();
#pragma unroll
  for (int q = 0; q < 4; ++q) {
    int i = q * 256 + t;
    int n = i >> 4, c4 = i & 15;         // n row, c4 = k-group
    *(ushort4*)(Wt + (size_t)(n0 + n) * DDIM + k0 + c4 * 4) =
        *(ushort4*)&tile[n][c4 * 4];
  }
}

// ---------------- Kernel 1: scores = ctx @ entT, argmax, gather -> X (bf16) ----------------
__global__ void __launch_bounds__(256)
k_score_argmax_gather(const float* __restrict__ context, unsigned short* __restrict__ X) {
  __shared__ unsigned short sA[LDIM * LSTR];
  __shared__ unsigned short sB[LDIM * LSTR];
  __shared__ int sIdx[LDIM];

  const int bk = blockIdx.x;
  const float* ctx = context + (size_t)(bk * 2 + 0) * LDIM * DDIM;
  const float* ent = context + (size_t)(bk * 2 + 1) * LDIM * DDIM;
  const int t = threadIdx.x, lane = t & 31, w = t >> 5;

  v8f acc[8];
#pragma unroll
  for (int i = 0; i < 8; ++i) acc[i] = v8f_zero();

  for (int dc = 0; dc < NCH; ++dc) {
    const int d0 = dc * CH;
#pragma unroll
    for (int q = 0; q < 8; ++q) {
      int i = q * 256 + t;               // 2048 float4
      int row = i >> 4, c4 = i & 15;
      float4 va = *(const float4*)(ctx + (size_t)row * DDIM + d0 + c4 * 4);
      float4 vb = *(const float4*)(ent + (size_t)row * DDIM + d0 + c4 * 4);
      ushort4 ua = { f2bf(va.x), f2bf(va.y), f2bf(va.z), f2bf(va.w) };
      ushort4 ub = { f2bf(vb.x), f2bf(vb.y), f2bf(vb.z), f2bf(vb.w) };
      *(ushort4*)(sA + row * LSTR + c4 * 4) = ua;
      *(ushort4*)(sB + row * LSTR + c4 * 4) = ub;
    }
    __syncthreads();
    const int r0 = w * 16;
#pragma unroll
    for (int kk = 0; kk < CH; kk += 32) {
      v16bf a = frag_a(sA, r0 + (lane & 15), kk, lane);
#pragma unroll
      for (int ct = 0; ct < 8; ++ct) {
        v16bf b = frag_b(sB, ct * 16 + (lane & 15), kk, lane);
        acc[ct] = __builtin_amdgcn_wmma_f32_16x16x32_bf16(
            false, a, false, b, (short)0, acc[ct], false, false);
      }
    }
    __syncthreads();
  }

  // per-row argmax over 128 columns (first-occurrence tie-break like jnp.argmax)
  float bv[8]; int bc[8];
#pragma unroll
  for (int j = 0; j < 8; ++j) { bv[j] = -3.4e38f; bc[j] = 0; }
#pragma unroll
  for (int ct = 0; ct < 8; ++ct) {
    int c = ct * 16 + (lane & 15);
#pragma unroll
    for (int j = 0; j < 8; ++j) {
      float v = acc[ct][j];
      if (v > bv[j] || (v == bv[j] && c < bc[j])) { bv[j] = v; bc[j] = c; }
    }
  }
#pragma unroll
  for (int m = 1; m < 16; m <<= 1) {
#pragma unroll
    for (int j = 0; j < 8; ++j) {
      float ov = __shfl_xor(bv[j], m, 32);
      int   oc = __shfl_xor(bc[j], m, 32);
      if (ov > bv[j] || (ov == bv[j] && oc < bc[j])) { bv[j] = ov; bc[j] = oc; }
    }
  }
  if ((lane & 15) == 0) {
    int rbase = w * 16 + ((lane >> 4) << 3);
#pragma unroll
    for (int j = 0; j < 8; ++j) sIdx[rbase + j] = bc[j];
  }
  __syncthreads();

  // gather + convert: X[pair-row] = (p==0) ? ctx[l] : ent[argmax[l]]
  unsigned short* Xb = X + (size_t)bk * 256 * DDIM;
  for (int q = 0; q < 192; ++q) {
    int i = q * 256 + t;                 // 49152 float4
    int pr = i / 192, c4 = i % 192;
    int l = pr >> 1, p = pr & 1;
    const float* src = p ? (ent + (size_t)sIdx[l] * DDIM) : (ctx + (size_t)l * DDIM);
    float4 v = *(const float4*)(src + c4 * 4);
    ushort4 u = { f2bf(v.x), f2bf(v.y), f2bf(v.z), f2bf(v.w) };
    *(ushort4*)(Xb + (size_t)pr * DDIM + c4 * 4) = u;
  }
}

// ---------------- Kernel 2: H1 = relu(X @ W1 + b1), bf16 out ----------------
// X and Wt1 are bf16 -> staging is pure 16B copies, async + double-buffered.
__global__ void __launch_bounds__(256)
k_gemm1(const unsigned short* __restrict__ X, const unsigned short* __restrict__ Wt1,
        const float* __restrict__ b1, unsigned short* __restrict__ H1) {
  __shared__ unsigned short sX[2][LDIM * LSTR];
  __shared__ unsigned short sW[2][LDIM * LSTR];
  const int nt = blockIdx.x * 128;
  const size_t rt = (size_t)blockIdx.y * 128;
  const int t = threadIdx.x, lane = t & 31, w = t >> 5;

  auto stage = [&](int buf, int dc) {
    const int d0 = dc * CH;
#pragma unroll
    for (int q = 0; q < 4; ++q) {
      int i = q * 256 + t;               // 1024 x 16B per tile
      int row = i >> 3, c8 = i & 7;
      cp16(sX[buf] + row * LSTR + c8 * 8, X + (rt + row) * DDIM + d0 + c8 * 8);
      cp16(sW[buf] + row * LSTR + c8 * 8, Wt1 + (size_t)(nt + row) * DDIM + d0 + c8 * 8);
    }
  };

  v8f acc[8];
#pragma unroll
  for (int i = 0; i < 8; ++i) acc[i] = v8f_zero();

  stage(0, 0);
  wait_async0();
  __syncthreads();
  for (int dc = 0; dc < NCH; ++dc) {
    const int cur = dc & 1;
    if (dc + 1 < NCH) stage(1 - cur, dc + 1);
#pragma unroll
    for (int kk = 0; kk < CH; kk += 32) {
      v16bf a = frag_a(sX[cur], w * 16 + (lane & 15), kk, lane);
#pragma unroll
      for (int ct = 0; ct < 8; ++ct) {
        v16bf b = frag_b(sW[cur], ct * 16 + (lane & 15), kk, lane);
        acc[ct] = __builtin_amdgcn_wmma_f32_16x16x32_bf16(
            false, a, false, b, (short)0, acc[ct], false, false);
      }
    }
    wait_async0();
    __syncthreads();
  }

  const int M = (lane >> 4) << 3;
#pragma unroll
  for (int ct = 0; ct < 8; ++ct) {
    int col = nt + ct * 16 + (lane & 15);
    float bv = b1[col];
#pragma unroll
    for (int j = 0; j < 8; ++j) {
      float h = acc[ct][j] + bv;
      h = h > 0.0f ? h : 0.0f;
      size_t gr = rt + w * 16 + M + j;
      H1[gr * DDIM + col] = f2bf(h);
    }
  }
}

// -------- Kernel 3: relu(H1 @ W2 + b2) contracted with W3, +b3, pair-sum -> out --------
__global__ void __launch_bounds__(256)
k_gemm2_out(const unsigned short* __restrict__ H1, const unsigned short* __restrict__ Wt2,
            const float* __restrict__ b2, const float* __restrict__ W3,
            const float* __restrict__ b3, float* __restrict__ out) {
  __shared__ unsigned short sX[2][LDIM * LSTR];
  __shared__ unsigned short sW[2][LDIM * LSTR];
  const size_t rt = (size_t)blockIdx.x * 128;
  const int t = threadIdx.x, lane = t & 31, w = t >> 5;

  float rowdot[8];
#pragma unroll
  for (int j = 0; j < 8; ++j) rowdot[j] = 0.0f;

  for (int cb = 0; cb < 6; ++cb) {
    const int nt = cb * 128;
    auto stage = [&](int buf, int dc) {
      const int d0 = dc * CH;
#pragma unroll
      for (int q = 0; q < 4; ++q) {
        int i = q * 256 + t;
        int row = i >> 3, c8 = i & 7;
        cp16(sX[buf] + row * LSTR + c8 * 8, H1 + (rt + row) * DDIM + d0 + c8 * 8);
        cp16(sW[buf] + row * LSTR + c8 * 8, Wt2 + (size_t)(nt + row) * DDIM + d0 + c8 * 8);
      }
    };

    v8f acc[8];
#pragma unroll
    for (int i = 0; i < 8; ++i) acc[i] = v8f_zero();

    stage(0, 0);
    wait_async0();
    __syncthreads();
    for (int dc = 0; dc < NCH; ++dc) {
      const int cur = dc & 1;
      if (dc + 1 < NCH) stage(1 - cur, dc + 1);
#pragma unroll
      for (int kk = 0; kk < CH; kk += 32) {
        v16bf a = frag_a(sX[cur], w * 16 + (lane & 15), kk, lane);
#pragma unroll
        for (int ct = 0; ct < 8; ++ct) {
          v16bf b = frag_b(sW[cur], ct * 16 + (lane & 15), kk, lane);
          acc[ct] = __builtin_amdgcn_wmma_f32_16x16x32_bf16(
              false, a, false, b, (short)0, acc[ct], false, false);
        }
      }
      wait_async0();
      __syncthreads();
    }
    // contract this col-block with W3 (bias + relu first)
#pragma unroll
    for (int ct = 0; ct < 8; ++ct) {
      int col = nt + ct * 16 + (lane & 15);
      float b2v = b2[col];
      float w3v = W3[col];
#pragma unroll
      for (int j = 0; j < 8; ++j) {
        float h = acc[ct][j] + b2v;
        h = h > 0.0f ? h : 0.0f;
        rowdot[j] += h * w3v;
      }
    }
  }
#pragma unroll
  for (int m = 1; m < 16; m <<= 1) {
#pragma unroll
    for (int j = 0; j < 8; ++j) rowdot[j] += __shfl_xor(rowdot[j], m, 32);
  }
  if ((lane & 15) == 0) {
    float b3v = b3[0];
    int rbase = w * 16 + ((lane >> 4) << 3);
#pragma unroll
    for (int m = 0; m < 4; ++m) {
      size_t gr = rt + rbase + 2 * m;           // pair rows (l,0),(l,1)
      out[gr >> 1] = rowdot[2 * m] + rowdot[2 * m + 1] + 2.0f * b3v;
    }
  }
}

extern "C" void kernel_launch(void* const* d_in, const int* in_sizes, int n_in,
                              void* d_out, int out_size, void* d_ws, size_t ws_size,
                              hipStream_t stream) {
  (void)in_sizes; (void)n_in; (void)out_size; (void)ws_size;
  const float* context = (const float*)d_in[0];
  const float* W1 = (const float*)d_in[1];
  const float* b1 = (const float*)d_in[2];
  const float* W2 = (const float*)d_in[3];
  const float* b2 = (const float*)d_in[4];
  const float* W3 = (const float*)d_in[5];
  const float* b3 = (const float*)d_in[6];

  unsigned short* X   = (unsigned short*)d_ws;            // 65536*768 bf16
  unsigned short* H1  = X  + (size_t)65536 * DDIM;        // 65536*768 bf16
  unsigned short* Wt1 = H1 + (size_t)65536 * DDIM;        // 768*768 bf16
  unsigned short* Wt2 = Wt1 + (size_t)DDIM * DDIM;        // 768*768 bf16
  float* out = (float*)d_out;                             // 32768 f32

  k_wtrans<<<144, 256, 0, stream>>>(W1, Wt1);
  k_wtrans<<<144, 256, 0, stream>>>(W2, Wt2);
  k_score_argmax_gather<<<NBK, 256, 0, stream>>>(context, X);
  k_gemm1<<<dim3(6, 512), 256, 0, stream>>>(X, Wt1, b1, H1);
  k_gemm2_out<<<512, 256, 0, stream>>>(H1, Wt2, b2, W3, b3, out);
}
